// GDMLTorchAssemble_5188320493598
// MI455X (gfx1250) — matrix operand order repaired and checked by reading, not compiled, not run on hardware
//
#include <hip/hip_runtime.h>
#include <hip/hip_bf16.h>

// ---------------------------------------------------------------------------
// GDML Matern-5/2 kernel-column assembly for MI455X (gfx1250, wave32, WMMA).
//
// Per training point n (one 256-thread workgroup):
//   Xd[p][d]   = q*(R_desc[n,d] - R_desc[j, perm_p(d)])          (LDS)
//   dist[p], exp_xs[p], e1[p]                                    (LDS reduce)
//   inner[p,i] = sum_d Xd[p][d] * J_p[d,i]        (J_p 6-sparse -> ds atomics)
//   diff_ab[d,i] = sum_p Xd[p][d]*ex[p]*inner[p,i] - sum_p J_p[d,i]*e1[p]
//                  (stored transposed Bt32[i][d] f32, converted to bf16)
//   A[i][d]    = Rd_full[n][d][i]                 (6-sparse fill, bf16)
//   k[n]       = A(96x448) * B(448x96)  via v_wmma_f32_16x16x32_bf16
//
// LDS budget (~284 KB of the 320 KB WGP pool, one WG per WGP):
//   [0,172032)        Bt32 f32 [96][448]   -- later first half reused as A bf16
//   [172032,258048)   Bt16 bf16 [96][448]
//   [258048,279552)   Xd   f32 [12][448]
//   [279552,284160)   inner f32 [12][96]
//   [284160,284304)   ss/ex/e1 (12 floats each)
// ---------------------------------------------------------------------------

#define NTRAIN 2000
#define NATOMS 30
#define DIMD   435
#define DPAD   448   // 14 * 32 (WMMA K-steps)
#define DIMI   90
#define IPAD   96    // 6 * 16 (WMMA tiles)
#define NPERM  12

typedef __attribute__((ext_vector_type(16))) __bf16 v16bf;
typedef __attribute__((ext_vector_type(8)))  __bf16 v8bf;
typedef __attribute__((ext_vector_type(8)))  float  v8f;

#define OFF_B16   (IPAD * DPAD * 4)                 // 172032
#define OFF_XD    (OFF_B16 + IPAD * DPAD * 2)       // 258048
#define OFF_INNER (OFF_XD + NPERM * DPAD * 4)       // 279552
#define OFF_SS    (OFF_INNER + NPERM * IPAD * 4)    // 284160
#define SMEM_SZ   (OFF_SS + 3 * NPERM * 4)          // 284304

__device__ __forceinline__ void inv_tril(int dd, int& r, int& c) {
    // invert dd = r*(r-1)/2 + c  (c < r), i.e. np.tril_indices(N,-1) order
    r = (int)((1.0f + sqrtf(8.0f * (float)dd + 1.0f)) * 0.5f);
    while (r * (r - 1) / 2 > dd) --r;
    while ((r + 1) * r / 2 <= dd) ++r;
    c = dd - r * (r - 1) / 2;
}

// ---------------------------------------------------------------------------
// Setup: permutation tables + point-j descriptor/Jacobian values (shared by
// all 2000 blocks).  rjdp[p][d] = R_desc[j, pd];  v3tab[p][d] = R_d_desc[j,pd,:]
// ptab packs pd | row_atom<<16 | col_atom<<24 for the PERMUTED pair index.
// rowcol packs the UNPERMUTED pair atoms for building Rd_full.
// ---------------------------------------------------------------------------
__global__ __launch_bounds__(256) void gdml_setup(
    const float* __restrict__ R_desc, const float* __restrict__ R_d_desc,
    const int* __restrict__ tpl, const int* __restrict__ jptr,
    float* __restrict__ rjdp, int* __restrict__ ptab,
    float* __restrict__ v3tab, int* __restrict__ rowcol)
{
    const int j = jptr[0];
    const int idx = blockIdx.x * blockDim.x + threadIdx.x;

    if (idx < DPAD) {
        int rc = 0;
        if (idx < DIMD) { int r, c; inv_tril(idx, r, c); rc = r | (c << 8); }
        rowcol[idx] = rc;
    }
    if (idx < NPERM * DPAD) {
        const int p = idx / DPAD, d = idx - p * DPAD;
        float rj = 0.f, v0 = 0.f, v1 = 0.f, v2 = 0.f;
        int pk = 0;
        if (d < DIMD) {
            const int pd = tpl[d * NPERM + p] % DIMD;  // tril_perms[p][d]
            int r, c; inv_tril(pd, r, c);
            pk = pd | (r << 16) | (c << 24);
            rj = R_desc[(size_t)j * DIMD + pd];
            const float* v = &R_d_desc[((size_t)j * DIMD + pd) * 3];
            v0 = v[0]; v1 = v[1]; v2 = v[2];
        }
        ptab[idx] = pk;
        rjdp[idx] = rj;
        float* o = &v3tab[(size_t)idx * 4];
        o[0] = v0; o[1] = v1; o[2] = v2; o[3] = 0.f;
    }
}

// ---------------------------------------------------------------------------
// Main: one workgroup per training point.
// ---------------------------------------------------------------------------
__global__ __launch_bounds__(256) void gdml_main(
    const float* __restrict__ R_desc, const float* __restrict__ R_d_desc,
    const float* __restrict__ rjdp, const int* __restrict__ ptab,
    const float* __restrict__ v3tab, const int* __restrict__ rowcol,
    float* __restrict__ out)
{
    extern __shared__ char smem[];
    float*  Bt32 = (float*)smem;                 // [IPAD][DPAD] f32
    __bf16* A16  = (__bf16*)smem;                // alias (used after convert)
    __bf16* Bt16 = (__bf16*)(smem + OFF_B16);    // [IPAD][DPAD] bf16
    float*  Xd   = (float*)(smem + OFF_XD);      // [NPERM][DPAD]
    float*  innr = (float*)(smem + OFF_INNER);   // [NPERM][IPAD]
    float*  ss   = (float*)(smem + OFF_SS);      // [NPERM] dist^2
    float*  exv  = ss + NPERM;                   // exp_xs
    float*  e1v  = ss + 2 * NPERM;               // exp_xs*(1+dist)

    const int n = blockIdx.x;
    const int tid = threadIdx.x;
    const float q = 0.22360679774997896f;        // sqrt(5)/SIG

    // ---- phase 0: zero accumulators ------------------------------------
    for (int i = tid; i < IPAD * DPAD; i += 256) Bt32[i] = 0.f;
    for (int i = tid; i < NPERM * IPAD; i += 256) innr[i] = 0.f;
    if (tid < NPERM) ss[tid] = 0.f;
    __syncthreads();

    // ---- phase 1: x_diffs + squared distances --------------------------
    for (int e = tid; e < NPERM * DPAD; e += 256) {
        const int p = e / DPAD, d = e - p * DPAD;
        float x = 0.f;
        if (d < DIMD) {
            x = q * (R_desc[(size_t)n * DIMD + d] - rjdp[e]);
            atomicAdd(&ss[p], x * x);
        }
        Xd[e] = x;
    }
    __syncthreads();

    // ---- phase 2: exp_xs, e1 -------------------------------------------
    if (tid < NPERM) {
        const float dist = sqrtf(ss[tid]);
        const float ex = __expf(-dist) * (q * q / 3.0f);
        exv[tid] = ex;
        e1v[tid] = ex * (1.0f + dist);
    }
    __syncthreads();

    // ---- phase 3: sparse scatter: inner[p,i] and -J_p*e1 into Bt32 -----
    for (int e = tid; e < NPERM * DIMD; e += 256) {
        const int p = e / DIMD, d = e - p * DIMD;
        const int pk = ptab[p * DPAD + d];
        const int ra = (pk >> 16) & 0xff, ca = (pk >> 24) & 0xff;
        const float xv = Xd[p * DPAD + d];
        const float sc = e1v[p];
        const float* v3 = &v3tab[(size_t)(p * DPAD + d) * 4];
#pragma unroll
        for (int c = 0; c < 3; ++c) {
            const float v = v3[c];
            atomicAdd(&innr[p * IPAD + 3 * ra + c],  xv * v);
            atomicAdd(&innr[p * IPAD + 3 * ca + c], -xv * v);
            atomicAdd(&Bt32[(3 * ra + c) * DPAD + d], -sc * v);
            atomicAdd(&Bt32[(3 * ca + c) * DPAD + d],  sc * v);
        }
    }
    __syncthreads();

    // ---- phase 4: rank-12 term into diff_ab ----------------------------
    for (int d = tid; d < DIMD; d += 256) {
        float xe[NPERM];
#pragma unroll
        for (int p = 0; p < NPERM; ++p) xe[p] = Xd[p * DPAD + d] * exv[p];
        for (int i = 0; i < DIMI; ++i) {
            float acc = 0.f;
#pragma unroll
            for (int p = 0; p < NPERM; ++p) acc += xe[p] * innr[p * IPAD + i];
            Bt32[i * DPAD + d] += acc;
        }
    }
    __syncthreads();

    // ---- phase 5: convert diff_ab to bf16 ------------------------------
    for (int e = tid; e < IPAD * DPAD; e += 256) Bt16[e] = (__bf16)Bt32[e];
    __syncthreads();

    // ---- phase 6: build A = Rd_full[n]^T (bf16, reuses Bt32 memory) ----
    for (int e = tid; e < IPAD * DPAD; e += 256) A16[e] = (__bf16)0.0f;
    __syncthreads();
    for (int d = tid; d < DIMD; d += 256) {
        const int rc = rowcol[d];
        const int ra = rc & 0xff, ca = (rc >> 8) & 0xff;
#pragma unroll
        for (int c = 0; c < 3; ++c) {
            const float v = R_d_desc[((size_t)n * DIMD + d) * 3 + c];
            A16[(3 * ra + c) * DPAD + d] = (__bf16)v;
            A16[(3 * ca + c) * DPAD + d] = (__bf16)(-v);
        }
    }
    __syncthreads();

    // ---- phase 7: k[n] = A(96x448) * B(448x96) via WMMA ---------------
    // A fragment (16x32 bf16): lane m=lane&15, half=lane>>4;
    //   regs 0..3 hold K = 8*half+0..7, regs 4..7 hold K = 16+8*half+0..7
    //   -> two 16B contiguous chunks per lane from A16[row][.]
    // B fragment (32x16 bf16): lane n=lane&15, half=lane>>4;
    //   same K order as A -> two 16B chunks from Bt16[col][.]
    const int wave = tid >> 5, lane = tid & 31;
    const int half = lane >> 4, l15 = lane & 15;
    for (int t = wave; t < 36; t += 8) {
        const int tm = t / 6, tn = t - tm * 6;
        v8f acc = {0.f, 0.f, 0.f, 0.f, 0.f, 0.f, 0.f, 0.f};
        const __bf16* arow = &A16[(tm * 16 + l15) * DPAD];
        const __bf16* brow = &Bt16[(tn * 16 + l15) * DPAD];
#pragma unroll
        for (int kk = 0; kk < DPAD / 32; ++kk) {
            const int ka = kk * 32 + 8 * half;
            const v8bf alo = *(const v8bf*)(arow + ka);
            const v8bf ahi = *(const v8bf*)(arow + ka + 16);
            const int kb = kk * 32 + 8 * half;
            const v8bf blo = *(const v8bf*)(brow + kb);
            const v8bf bhi = *(const v8bf*)(brow + kb + 16);
            v16bf af, bv;
#pragma unroll
            for (int e2 = 0; e2 < 8; ++e2) {
                af[e2] = alo[e2]; af[e2 + 8] = ahi[e2];
                bv[e2] = blo[e2]; bv[e2 + 8] = bhi[e2];
            }
            acc = __builtin_amdgcn_wmma_f32_16x16x32_bf16(
                false, af, false, bv, (short)0, acc, false, false);
        }
        // C/D layout: VGPR r -> row r + 8*half, lane&15 -> column
#pragma unroll
        for (int r = 0; r < 8; ++r) {
            const int m = tm * 16 + r + 8 * half;
            const int col = tn * 16 + l15;
            if (m < DIMI && col < DIMI)
                out[((size_t)n * DIMI + m) * DIMI + col] = acc[r];
        }
    }
}

// ---------------------------------------------------------------------------
extern "C" void kernel_launch(void* const* d_in, const int* in_sizes, int n_in,
                              void* d_out, int out_size, void* d_ws, size_t ws_size,
                              hipStream_t stream) {
    (void)in_sizes; (void)n_in; (void)out_size; (void)ws_size;
    const float* R_desc   = (const float*)d_in[0];   // [2000,435]
    const float* R_d_desc = (const float*)d_in[1];   // [2000,435,3]
    const int*   tpl      = (const int*)d_in[2];     // [5220]
    const int*   jptr     = (const int*)d_in[3];     // scalar
    float* out = (float*)d_out;                      // [2000*90,90]

    char* ws = (char*)d_ws;
    float* rjdp   = (float*)ws;                               // 12*448 f32
    int*   ptab   = (int*)(ws + NPERM * DPAD * 4);            // 12*448 i32
    float* v3tab  = (float*)(ws + 2 * NPERM * DPAD * 4);      // 12*448*4 f32
    int*   rowcol = (int*)(ws + 2 * NPERM * DPAD * 4 + NPERM * DPAD * 16);

    gdml_setup<<<(NPERM * DPAD + 255) / 256, 256, 0, stream>>>(
        R_desc, R_d_desc, tpl, jptr, rjdp, ptab, v3tab, rowcol);

    // 284304 B dynamic LDS: one workgroup per WGP (320 KB pool on CDNA5).
    gdml_main<<<NTRAIN, 256, SMEM_SZ, stream>>>(
        R_desc, R_d_desc, rjdp, ptab, v3tab, rowcol, out);
}
